// AttnDecoderRNN_74526272520561
// MI455X (gfx1250) — matrix operand order repaired and checked by reading, not compiled, not run on hardware
//
#include <hip/hip_runtime.h>
#include <math.h>

#define VOCAB  32000
#define EMBED  1024
#define HIDDEN 2048
#define MAXLEN 4096

// ---------------------------------------------------------------- WMMA types
typedef __attribute__((ext_vector_type(16))) __bf16 v16bf;
typedef __attribute__((ext_vector_type(8)))  __bf16 v8bf;
typedef __attribute__((ext_vector_type(8)))  float  v8f;

__device__ __forceinline__ unsigned bf16_rne(float f) {
    unsigned u = __float_as_uint(f);
    return (u + 0x7FFFu + ((u >> 16) & 1u)) >> 16;
}
__device__ __forceinline__ unsigned pack_bf16(float a, float b) {
    return bf16_rne(a) | (bf16_rne(b) << 16);
}
__device__ __forceinline__ float sigmoidf_(float v) { return 1.0f / (1.0f + expf(-v)); }

// ---------------------------------------------------------------------------
// score_c GEMM: T = tanh(enc(4096x2048) @ Wc^T(2048x2048) + b), score_c = T @ h_new
// bf16 WMMA with f32 accumulate; tanh+dot fused in epilogue (T never stored).
// Block tile 128(M) x 64(N), BK=32. 8 waves as 4x2, each wave 2x2 WMMA tiles.
// ---------------------------------------------------------------------------
__global__ __launch_bounds__(256) void score_c_gemm_kernel(
    const float* __restrict__ enc, const float* __restrict__ Wc,
    const float* __restrict__ Wc_b, const float* __restrict__ h_new,
    float* __restrict__ score_c)
{
    __shared__ __align__(16) unsigned As[128 * 16]; // 128 rows x 32 K (bf16 pairs)
    __shared__ __align__(16) unsigned Bs[64 * 16];  // 64 cols  x 32 K (bf16 pairs)

    const int tid  = threadIdx.x;
    const int lane = tid & 31;
    const int wave = tid >> 5;
    const int wm   = wave >> 1;   // 0..3 -> M position
    const int wn   = wave & 1;    // 0..1 -> N position
    const int i0   = blockIdx.y * 128;
    const int j0   = blockIdx.x * 64;

    v8f acc[2][2] = {};

    for (int k0 = 0; k0 < HIDDEN; k0 += 32) {
        // ---- stage A tile (convert f32 -> packed bf16) : 2048 words, 8/thread
        #pragma unroll
        for (int e = 0; e < 8; ++e) {
            int p = e * 256 + tid;
            int row = p >> 4, wp = p & 15;
            const float* s = enc + (size_t)(i0 + row) * HIDDEN + k0 + wp * 2;
            As[row * 16 + wp] = pack_bf16(s[0], s[1]);
        }
        // ---- stage B tile : 1024 words, 4/thread (B col j = Wc row j, K-contig)
        #pragma unroll
        for (int e = 0; e < 4; ++e) {
            int p = e * 256 + tid;
            int row = p >> 4, wp = p & 15;
            const float* s = Wc + (size_t)(j0 + row) * HIDDEN + k0 + wp * 2;
            Bs[row * 16 + wp] = pack_bf16(s[0], s[1]);
        }
        __syncthreads();

        // ---- A fragments: lane<16 holds K {0..7,16..23}; lane>=16 K {8..15,24..31}
        v16bf afrag[2], bfrag[2];
        #pragma unroll
        for (int mt = 0; mt < 2; ++mt) {
            int row = wm * 32 + mt * 16 + (lane & 15);
            int kw  = (lane < 16) ? 0 : 4;
            v8bf lo = *(const v8bf*)&As[row * 16 + kw];
            v8bf hi = *(const v8bf*)&As[row * 16 + kw + 8];
            afrag[mt] = __builtin_shufflevector(lo, hi, 0,1,2,3,4,5,6,7,8,9,10,11,12,13,14,15);
        }
        // ---- B fragments: lane&15 = column; lane<16 K {0..7,16..23}; lane>=16 K {8..15,24..31}
        #pragma unroll
        for (int nt = 0; nt < 2; ++nt) {
            int col = wn * 32 + nt * 16 + (lane & 15);
            int kw  = (lane < 16) ? 0 : 4;
            v8bf lo = *(const v8bf*)&Bs[col * 16 + kw];
            v8bf hi = *(const v8bf*)&Bs[col * 16 + kw + 8];
            bfrag[nt] = __builtin_shufflevector(lo, hi, 0,1,2,3,4,5,6,7,8,9,10,11,12,13,14,15);
        }
        #pragma unroll
        for (int mt = 0; mt < 2; ++mt)
            #pragma unroll
            for (int nt = 0; nt < 2; ++nt)
                acc[mt][nt] = __builtin_amdgcn_wmma_f32_16x16x32_bf16(
                    false, afrag[mt], false, bfrag[nt], (short)0, acc[mt][nt], false, false);
        __syncthreads();
    }

    // ---- epilogue: tanh(+bias) * h_new, reduce 16 N-lanes per row, atomic add
    #pragma unroll
    for (int nt = 0; nt < 2; ++nt) {
        int j = j0 + wn * 32 + nt * 16 + (lane & 15);
        float bj = Wc_b[j];
        float hj = h_new[j];
        #pragma unroll
        for (int mt = 0; mt < 2; ++mt) {
            #pragma unroll
            for (int r = 0; r < 8; ++r) {
                float v = tanhf(acc[mt][nt][r] + bj) * hj;
                v += __shfl_xor(v, 1, 16);
                v += __shfl_xor(v, 2, 16);
                v += __shfl_xor(v, 4, 16);
                v += __shfl_xor(v, 8, 16);
                if ((lane & 15) == 0) {
                    int i = i0 + wm * 32 + mt * 16 + r + ((lane < 16) ? 0 : 8);
                    atomicAdd(&score_c[i], v);
                }
            }
        }
    }
}

// ----------------------------------------------------------- generic matvec
// y[r] = act(dot(W[r,:], x) + b[r]); one wave32 per row, float4 coalesced.
__global__ __launch_bounds__(256) void matvec_kernel(
    const float* __restrict__ W, const float* __restrict__ x,
    const float* __restrict__ bias, float* __restrict__ y,
    int rows, int cols, int act)
{
    int row  = (blockIdx.x * 256 + threadIdx.x) >> 5;
    int lane = threadIdx.x & 31;
    if (row >= rows) return;
    const float4* Wr = (const float4*)(W + (size_t)row * cols);
    const float4* xv = (const float4*)x;
    int nv = cols >> 2;
    float acc = 0.0f;
    for (int c = lane; c < nv; c += 32) {
        __builtin_prefetch(&Wr[c + 64], 0, 0);
        float4 w4 = Wr[c];
        float4 x4 = xv[c];
        acc = fmaf(w4.x, x4.x, acc);
        acc = fmaf(w4.y, x4.y, acc);
        acc = fmaf(w4.z, x4.z, acc);
        acc = fmaf(w4.w, x4.w, acc);
    }
    #pragma unroll
    for (int m = 16; m >= 1; m >>= 1) acc += __shfl_xor(acc, m, 32);
    if (lane == 0) {
        float v = acc + bias[row];
        if (act == 1) v = fmaxf(v, 0.0f);
        y[row] = v;
    }
}

// --------------------------------- sel_reading[h] += sel[i]*trunc(enc[i,h])
__global__ __launch_bounds__(256) void sel_reading_kernel(
    const float* __restrict__ enc, const int* __restrict__ input,
    const int* __restrict__ input_seq, const float* __restrict__ pre_prob,
    float* __restrict__ out)
{
    __shared__ float s_w[256];
    int tid = threadIdx.x;
    int col = blockIdx.x * 256 + tid;
    int r0  = blockIdx.y * 256;
    int tok = *input;
    s_w[tid] = (input_seq[r0 + tid] == tok) ? pre_prob[r0 + tid] : 0.0f;
    __syncthreads();
    float acc = 0.0f;
    for (int r = 0; r < 256; ++r) {
        float w = s_w[r];                 // uniform branch -> skips whole rows
        if (w != 0.0f) acc += w * truncf(enc[(size_t)(r0 + r) * HIDDEN + col]);
    }
    if (acc != 0.0f) atomicAdd(&out[col], acc);
}

// --------------------------------- attn_applied[h] += w[i]*enc[i,h]
__global__ __launch_bounds__(256) void weighted_rowsum_kernel(
    const float* __restrict__ enc, const float* __restrict__ w,
    float* __restrict__ out)
{
    __shared__ float s_w[256];
    int tid = threadIdx.x;
    int col = blockIdx.x * 256 + tid;
    int r0  = blockIdx.y * 256;
    s_w[tid] = w[r0 + tid];
    __syncthreads();
    float acc = 0.0f;
    for (int r = 0; r < 256; ++r)
        acc += s_w[r] * enc[(size_t)(r0 + r) * HIDDEN + col];
    atomicAdd(&out[col], acc);
}

// --------------------------------- build [embedded ; h0] and comb_in prefix
__global__ void gather_concat_kernel(
    const float* __restrict__ emb, const int* __restrict__ input,
    const float* __restrict__ hidden, float* __restrict__ attn_in,
    float* __restrict__ comb_in)
{
    int t = blockIdx.x * 256 + threadIdx.x;
    if (t < EMBED) {
        float e = emb[(size_t)(*input) * EMBED + t];
        attn_in[t] = e;
        comb_in[t] = e;
    } else if (t < EMBED + HIDDEN) {
        attn_in[t] = hidden[t - EMBED];
    }
}

// --------------------------------- softmax over 4096 attn logits
__global__ __launch_bounds__(1024) void softmax4096_kernel(
    const float* __restrict__ logits, float* __restrict__ out)
{
    __shared__ float red[1024];
    int tid = threadIdx.x;
    float m = -INFINITY;
    for (int i = tid; i < MAXLEN; i += 1024) m = fmaxf(m, logits[i]);
    red[tid] = m; __syncthreads();
    for (int s = 512; s > 0; s >>= 1) {
        if (tid < s) red[tid] = fmaxf(red[tid], red[tid + s]);
        __syncthreads();
    }
    m = red[0]; __syncthreads();
    float sum = 0.0f;
    for (int i = tid; i < MAXLEN; i += 1024) sum += expf(logits[i] - m);
    red[tid] = sum; __syncthreads();
    for (int s = 512; s > 0; s >>= 1) {
        if (tid < s) red[tid] += red[tid + s];
        __syncthreads();
    }
    float inv = 1.0f / red[0];
    for (int i = tid; i < MAXLEN; i += 1024) out[i] = expf(logits[i] - m) * inv;
}

// --------------------------------- GRU gates
__global__ void gru_kernel(const float* __restrict__ gx, const float* __restrict__ gh,
                           const float* __restrict__ hidden, float* __restrict__ h_new)
{
    int t = blockIdx.x * 256 + threadIdx.x;
    if (t >= HIDDEN) return;
    float rg = sigmoidf_(gx[t] + gh[t]);
    float zg = sigmoidf_(gx[HIDDEN + t] + gh[HIDDEN + t]);
    float ng = tanhf(gx[2 * HIDDEN + t] + rg * gh[2 * HIDDEN + t]);
    h_new[t] = (1.0f - zg) * ng + zg * hidden[t];
}

// --------------------------------- log_softmax over [score_g ; score_c]
__global__ __launch_bounds__(1024) void log_softmax_kernel(
    const float* __restrict__ score_g, const float* __restrict__ score_c,
    float* __restrict__ out1, float* __restrict__ prob_c)
{
    __shared__ float red[1024];
    const int N = VOCAB + MAXLEN;
    int tid = threadIdx.x;
    float m = -INFINITY;
    for (int i = tid; i < N; i += 1024) {
        float v = (i < VOCAB) ? score_g[i] : score_c[i - VOCAB];
        m = fmaxf(m, v);
    }
    red[tid] = m; __syncthreads();
    for (int s = 512; s > 0; s >>= 1) {
        if (tid < s) red[tid] = fmaxf(red[tid], red[tid + s]);
        __syncthreads();
    }
    m = red[0]; __syncthreads();
    float sum = 0.0f;
    for (int i = tid; i < N; i += 1024) {
        float v = (i < VOCAB) ? score_g[i] : score_c[i - VOCAB];
        sum += expf(v - m);
    }
    red[tid] = sum; __syncthreads();
    for (int s = 512; s > 0; s >>= 1) {
        if (tid < s) red[tid] += red[tid + s];
        __syncthreads();
    }
    float logZ = m + logf(red[0]);
    for (int i = tid; i < N; i += 1024) {
        if (i < VOCAB) out1[i] = score_g[i] - logZ;          // prob_g
        else           prob_c[i - VOCAB] = score_c[i - VOCAB] - logZ;
    }
}

// --------------------------------- output1 += segment_sum(prob_c, input_seq)
__global__ void segsum_kernel(const int* __restrict__ input_seq,
                              const float* __restrict__ prob_c,
                              float* __restrict__ out1)
{
    int i = blockIdx.x * 256 + threadIdx.x;
    if (i < MAXLEN) atomicAdd(&out1[input_seq[i]], prob_c[i]);
}

// ===========================================================================
extern "C" void kernel_launch(void* const* d_in, const int* in_sizes, int n_in,
                              void* d_out, int out_size, void* d_ws, size_t ws_size,
                              hipStream_t stream)
{
    (void)in_sizes; (void)n_in; (void)out_size; (void)ws_size;

    const int*   input     = (const int*)  d_in[0];
    const float* hidden    = (const float*)d_in[1];   // h0 (2048)
    const float* enc       = (const float*)d_in[2];   // 4096 x 2048
    const int*   input_seq = (const int*)  d_in[3];
    const float* pre_prob  = (const float*)d_in[4];
    const float* emb       = (const float*)d_in[5];   // 32000 x 1024
    const float* attn_W    = (const float*)d_in[6];   // 4096 x 3072
    const float* attn_b    = (const float*)d_in[7];
    const float* comb_W    = (const float*)d_in[8];   // 2048 x 3072
    const float* comb_b    = (const float*)d_in[9];
    const float* W_ih      = (const float*)d_in[10];  // 6144 x 4096
    const float* W_hh      = (const float*)d_in[11];  // 6144 x 2048
    const float* b_ih      = (const float*)d_in[12];
    const float* b_hh      = (const float*)d_in[13];
    const float* Wo_W      = (const float*)d_in[14];  // 32000 x 2048
    const float* Wo_b      = (const float*)d_in[15];
    const float* Wc_W      = (const float*)d_in[16];  // 2048 x 2048
    const float* Wc_b      = (const float*)d_in[17];

    // workspace layout (f32)
    float* fws         = (float*)d_ws;
    float* attn_in     = fws;                   // 3072  [embedded ; h0]
    float* attn_logits = attn_in + 3072;        // 4096
    float* comb_in     = attn_logits + 4096;    // 3072  [embedded ; attn_applied]
    float* xvec        = comb_in + 3072;        // 4096  [out_c ; sel_reading]
    float* gx          = xvec + 4096;           // 6144
    float* gh          = gx + 6144;             // 6144
    float* score_g     = gh + 6144;             // 32000
    float* score_c     = score_g + VOCAB;       // 4096

    // output layout (f32): output1 | h_new | attn_weights | prob_c
    float* out1   = (float*)d_out;              // 32000
    float* h_new  = out1 + VOCAB;               // 2048
    float* attn_w = h_new + HIDDEN;             // 4096
    float* prob_c = attn_w + MAXLEN;            // 4096

    // zero accumulators (graph-capture-safe memset nodes)
    hipMemsetAsync(xvec + HIDDEN,   0, HIDDEN * sizeof(float), stream); // sel_reading
    hipMemsetAsync(comb_in + EMBED, 0, HIDDEN * sizeof(float), stream); // attn_applied
    hipMemsetAsync(score_c,         0, MAXLEN * sizeof(float), stream);

    // 1) embedded gather + concat inputs
    gather_concat_kernel<<<12, 256, 0, stream>>>(emb, input, hidden, attn_in, comb_in);
    // 2) sel_reading = sel @ trunc(enc)  -> xvec[2048:4096]
    sel_reading_kernel<<<dim3(8, 16), 256, 0, stream>>>(enc, input, input_seq, pre_prob,
                                                        xvec + HIDDEN);
    // 3) attn logits + softmax -> attn_weights (output)
    matvec_kernel<<<512, 256, 0, stream>>>(attn_W, attn_in, attn_b, attn_logits,
                                           MAXLEN, EMBED + HIDDEN, 0);
    softmax4096_kernel<<<1, 1024, 0, stream>>>(attn_logits, attn_w);
    // 4) attn_applied = attn_weights @ enc -> comb_in[1024:3072]
    weighted_rowsum_kernel<<<dim3(8, 16), 256, 0, stream>>>(enc, attn_w, comb_in + EMBED);
    // 5) out_c = relu(comb_W @ comb_in + b) -> xvec[0:2048]
    matvec_kernel<<<256, 256, 0, stream>>>(comb_W, comb_in, comb_b, xvec,
                                           HIDDEN, EMBED + HIDDEN, 1);
    // 6) GRU pre-activations
    matvec_kernel<<<768, 256, 0, stream>>>(W_ih, xvec, b_ih, gx, 3 * HIDDEN, 2 * HIDDEN * 2 / 2 + 2048, 0); // 6144 x 6144
    matvec_kernel<<<768, 256, 0, stream>>>(W_hh, hidden, b_hh, gh, 3 * HIDDEN, HIDDEN, 0);
    // 7) gates -> h_new (output)
    gru_kernel<<<8, 256, 0, stream>>>(gx, gh, hidden, h_new);
    // 8) score_g = Wo_W @ h_new + Wo_b
    matvec_kernel<<<4000, 256, 0, stream>>>(Wo_W, h_new, Wo_b, score_g, VOCAB, HIDDEN, 0);
    // 9) score_c via bf16 WMMA GEMM, fused tanh + dot(h_new)
    score_c_gemm_kernel<<<dim3(HIDDEN / 64, MAXLEN / 128), 256, 0, stream>>>(
        enc, Wc_W, Wc_b, h_new, score_c);
    // 10) log_softmax over concat -> prob_g (into out1) and prob_c (output)
    log_softmax_kernel<<<1, 1024, 0, stream>>>(score_g, score_c, out1, prob_c);
    // 11) output1 = prob_g + segment_sum(prob_c, input_seq)
    segsum_kernel<<<16, 256, 0, stream>>>(input_seq, prob_c, out1);
}
